// JointFeatureExtractor_56401510531285
// MI455X (gfx1250) — matrix operand order repaired and checked by reading, not compiled, not run on hardware
//
#include <hip/hip_runtime.h>

typedef __attribute__((ext_vector_type(16))) _Float16 v16h;
typedef __attribute__((ext_vector_type(8)))  _Float16 v8h;
typedef __attribute__((ext_vector_type(8)))  float    v8f;

#define JNT 17
#define LSJ 5
#define RSJ 6
#define LHJ 11
#define RHJ 12

#define TPB 256
#define AH_STRIDE 136   // halves: 128 cols + 8 pad (272B rows, 16B aligned, bank-spread)
#define W1_STRIDE 72    // halves: 64 cols + 8 pad (144B rows)
#define W2_STRIDE 136

struct __align__(16) SMem {
  _Float16 AH[TPB * AH_STRIDE];   // 69632 B: feats (cols 0..63) then h (cols 0..127)
  _Float16 W1[128 * W1_STRIDE];   // 18432 B: p1_W^T [n][k], k padded 60->64
  _Float16 W2[64 * W2_STRIDE];    // 17408 B: p2_W^T [n][k], k=128
  float ln1g[128], ln1b[128], p1b[128];
  float ln2g[64],  ln2b[64],  p2b[64];
  float je[JNT * 16];
  float posW[32], posb[16];
  float velW[16], velb[8];
  float angb[8];
  float relW[16], relb[8];
  float confW[4], confb[4];
  float stK[JNT][4];   // offx, offy, sx, sy for keypoints
  float stP[JNT][4];   // same for prev_keypoints
  float tc[JNT][2];    // torso center of normalized kp
};

__device__ __forceinline__ float clamp5(float v) {
  return fminf(fmaxf(v, -5.0f), 5.0f);
}

// torso-centric / bbox normalization parameters for one body
__device__ __forceinline__ void body_stats(const float* __restrict__ kp, float st[4]) {
  float mnx = 3.0e38f, mny = 3.0e38f, mxx = -3.0e38f, mxy = -3.0e38f;
  float jx[4], jy[4], jc[4];  // LS, RS, LH, RH
#pragma unroll
  for (int j = 0; j < JNT; ++j) {
    float x = kp[j * 3 + 0], y = kp[j * 3 + 1], c = kp[j * 3 + 2];
    mnx = fminf(mnx, x); mxx = fmaxf(mxx, x);
    mny = fminf(mny, y); mxy = fmaxf(mxy, y);
    if (j == LSJ) { jx[0] = x; jy[0] = y; jc[0] = c; }
    if (j == RSJ) { jx[1] = x; jy[1] = y; jc[1] = c; }
    if (j == LHJ) { jx[2] = x; jy[2] = y; jc[2] = c; }
    if (j == RHJ) { jx[3] = x; jy[3] = y; jc[3] = c; }
  }
  float wl = jc[0] > 0.f ? 1.f : 0.f, wr = jc[1] > 0.f ? 1.f : 0.f;
  float iw = 1.f / fmaxf(wl + wr, 1.f);
  float smx = (wl * jx[0] + wr * jx[1]) * iw, smy = (wl * jy[0] + wr * jy[1]) * iw;
  float hl = jc[2] > 0.f ? 1.f : 0.f, hr = jc[3] > 0.f ? 1.f : 0.f;
  float ih = 1.f / fmaxf(hl + hr, 1.f);
  float hmx = (hl * jx[2] + hr * jx[3]) * ih, hmy = (hl * jy[2] + hr * jy[3]) * ih;
  bool ut = ((wl + wr) > 0.f) && ((hl + hr) > 0.f);
  if (!ut) { smx = smy = hmx = hmy = 0.f; }
  float cx = (smx + hmx) * 0.5f, cy = (smy + hmy) * 0.5f;
  float dx = smx - hmx, dy = smy - hmy;
  float h = sqrtf(dx * dx + dy * dy);
  bool valid = ut && (h >= 10.0f);
  if (valid) {
    st[0] = cx; st[1] = cy; st[2] = 1.f / (h + 1e-6f); st[3] = st[2];
  } else {
    st[0] = mnx; st[1] = mny;
    st[2] = 1.f / (mxx - mnx + 1e-6f);
    st[3] = 1.f / (mxy - mny + 1e-6f);
  }
}

__global__ __launch_bounds__(TPB) void jfe_kernel(
    const float* __restrict__ kp, const float* __restrict__ pkp,
    const float* __restrict__ joint_emb,
    const float* __restrict__ pos_W, const float* __restrict__ pos_b,
    const float* __restrict__ vel_W, const float* __restrict__ vel_b,
    const float* __restrict__ angvel_b,
    const float* __restrict__ rel_W, const float* __restrict__ rel_b,
    const float* __restrict__ conf_W, const float* __restrict__ conf_b,
    const float* __restrict__ p1_W, const float* __restrict__ p1_b,
    const float* __restrict__ ln1_g, const float* __restrict__ ln1_b,
    const float* __restrict__ p2_W, const float* __restrict__ p2_b,
    const float* __restrict__ ln2_g, const float* __restrict__ ln2_b,
    float* __restrict__ out, int BJ) {
  __shared__ SMem sm;
  const int tid = threadIdx.x;
  const int t0 = blockIdx.x * TPB;

  // ---------------- phase 0: stage weights / params into LDS ----------------
  for (int i = tid; i < 128 * 64; i += TPB) {
    int n = i >> 6, k = i & 63;
    sm.W1[n * W1_STRIDE + k] = (k < 60) ? (_Float16)p1_W[k * 128 + n] : (_Float16)0.f;
  }
  for (int i = tid; i < 64 * 128; i += TPB) {
    int n = i >> 7, k = i & 127;
    sm.W2[n * W2_STRIDE + k] = (_Float16)p2_W[k * 64 + n];
  }
  for (int i = tid; i < 128; i += TPB) { sm.ln1g[i] = ln1_g[i]; sm.ln1b[i] = ln1_b[i]; sm.p1b[i] = p1_b[i]; }
  for (int i = tid; i < 64;  i += TPB) { sm.ln2g[i] = ln2_g[i]; sm.ln2b[i] = ln2_b[i]; sm.p2b[i] = p2_b[i]; }
  for (int i = tid; i < JNT * 16; i += TPB) sm.je[i] = joint_emb[i];
  if (tid < 32) sm.posW[tid] = pos_W[tid];
  if (tid < 16) { sm.posb[tid] = pos_b[tid]; sm.velW[tid] = vel_W[tid]; sm.relW[tid] = rel_W[tid]; }
  if (tid < 8)  { sm.velb[tid] = vel_b[tid]; sm.angb[tid] = angvel_b[tid]; sm.relb[tid] = rel_b[tid]; }
  if (tid < 4)  { sm.confW[tid] = conf_W[tid]; sm.confb[tid] = conf_b[tid]; }

  // per-body normalization stats (<= 17 bodies touched per 256-token block)
  const int nBodies = (BJ + JNT - 1) / JNT;
  const int bFirst = t0 / JNT;
  int bLast = (t0 + TPB - 1) / JNT;
  if (bLast > nBodies - 1) bLast = nBodies - 1;
  const int nb = bLast - bFirst + 1;
  if (tid < nb) {
    int b = bFirst + tid;
    float stk[4], stp[4];
    body_stats(kp  + (size_t)b * (JNT * 3), stk);
    body_stats(pkp + (size_t)b * (JNT * 3), stp);
    const float* kb = kp + (size_t)b * (JNT * 3);
    float tx = 0.f, ty = 0.f;
    const int idx4[4] = {LSJ, RSJ, LHJ, RHJ};
#pragma unroll
    for (int q = 0; q < 4; ++q) {
      float x = kb[idx4[q] * 3 + 0], y = kb[idx4[q] * 3 + 1];
      tx += clamp5((x - stk[0]) * stk[2]);
      ty += clamp5((y - stk[1]) * stk[3]);
    }
#pragma unroll
    for (int q = 0; q < 4; ++q) { sm.stK[tid][q] = stk[q]; sm.stP[tid][q] = stp[q]; }
    sm.tc[tid][0] = tx * 0.25f;
    sm.tc[tid][1] = ty * 0.25f;
  }
  __syncthreads();

  // ---------------- phase 1: per-token 60-dim features -> LDS (f16, K padded to 64)
  {
    int t = t0 + tid;
    float f[64];
    if (t < BJ) {
      int b = t / JNT;
      int j = t - b * JNT;
      int li = b - bFirst;
      const float* ktp = kp  + (size_t)t * 3;
      const float* ptp = pkp + (size_t)t * 3;
      float x = ktp[0], y = ktp[1], c = ktp[2];
      float px = ptp[0], py = ptp[1];
      float nx  = clamp5((x  - sm.stK[li][0]) * sm.stK[li][2]);
      float ny  = clamp5((y  - sm.stK[li][1]) * sm.stK[li][3]);
      float pnx = clamp5((px - sm.stP[li][0]) * sm.stP[li][2]);
      float pny = clamp5((py - sm.stP[li][1]) * sm.stP[li][3]);
      float vx = nx - pnx, vy = ny - pny;
      float rx = nx - sm.tc[li][0], ry = ny - sm.tc[li][1];
#pragma unroll
      for (int q = 0; q < 16; ++q) f[q] = sm.je[j * 16 + q];
#pragma unroll
      for (int q = 0; q < 16; ++q) f[16 + q] = nx * sm.posW[q] + ny * sm.posW[16 + q] + sm.posb[q];
#pragma unroll
      for (int q = 0; q < 8; ++q) f[32 + q] = vx * sm.velW[q] + vy * sm.velW[8 + q] + sm.velb[q];
#pragma unroll
      for (int q = 0; q < 8; ++q) f[40 + q] = sm.angb[q];
#pragma unroll
      for (int q = 0; q < 8; ++q) f[48 + q] = rx * sm.relW[q] + ry * sm.relW[8 + q] + sm.relb[q];
#pragma unroll
      for (int q = 0; q < 4; ++q) f[56 + q] = c * sm.confW[q] + sm.confb[q];
      f[60] = f[61] = f[62] = f[63] = 0.f;
    } else {
#pragma unroll
      for (int q = 0; q < 64; ++q) f[q] = 0.f;
    }
    _Float16* ar = &sm.AH[tid * AH_STRIDE];
#pragma unroll
    for (int q = 0; q < 64; ++q) ar[q] = (_Float16)f[q];
  }
  __syncthreads();

  // ---------------- phase 2: WMMA GEMMs ----------------
  const int wave  = tid >> 5;
  const int lane  = tid & 31;
  const int laneN = lane & 15;
  const bool hiHalf = lane >= 16;
  const int aKoff = hiHalf ? 8 : 0;    // A frag: K {0..7,16..23} | {8..15,24..31}
  const int bKoff = hiHalf ? 8 : 0;    // B frag: K {0..7,16..23} | {8..15,24..31}
  const int rOff  = hiHalf ? 8 : 0;    // D frag: lanes>=16 hold rows r+8

  float g1L[8], b1L[8], p1L[8];
#pragma unroll
  for (int nt = 0; nt < 8; ++nt) {
    int n = nt * 16 + laneN;
    g1L[nt] = sm.ln1g[n]; b1L[nt] = sm.ln1b[n]; p1L[nt] = sm.p1b[n];
  }
  float g2L[4], b2L[4], p2L[4];
#pragma unroll
  for (int nt = 0; nt < 4; ++nt) {
    int n = nt * 16 + laneN;
    g2L[nt] = sm.ln2g[n]; b2L[nt] = sm.ln2b[n]; p2L[nt] = sm.p2b[n];
  }

  union V16 { v16h v; v8h h[2]; };

  // ---- GEMM1: feats(16x64) @ W1 -> 16x128, LN1 + ReLU, write h back to AH ----
#pragma unroll
  for (int mt = 0; mt < 2; ++mt) {
    int mrow = wave * 32 + mt * 16 + laneN;
    V16 a[2];
#pragma unroll
    for (int kt = 0; kt < 2; ++kt) {
      const _Float16* base = &sm.AH[mrow * AH_STRIDE + kt * 32 + aKoff];
      a[kt].h[0] = *(const v8h*)base;
      a[kt].h[1] = *(const v8h*)(base + 16);
    }
    v8f acc[8];
#pragma unroll
    for (int nt = 0; nt < 8; ++nt) {
      v8f c = {};
#pragma unroll
      for (int kt = 0; kt < 2; ++kt) {
        int n = nt * 16 + laneN;
        const _Float16* wb = &sm.W1[n * W1_STRIDE + kt * 32 + bKoff];
        V16 bf;
        bf.h[0] = *(const v8h*)wb;
        bf.h[1] = *(const v8h*)(wb + 16);
        c = __builtin_amdgcn_wmma_f32_16x16x32_f16(false, a[kt].v, false, bf.v,
                                                   (short)0, c, false, false);
      }
      acc[nt] = c;
    }
    // LN1 (+bias, +ReLU); row r data: lanes 0-15 = row r, lanes 16-31 = row r+8
#pragma unroll
    for (int r = 0; r < 8; ++r) {
      float vals[8];
      float s = 0.f, q = 0.f;
#pragma unroll
      for (int nt = 0; nt < 8; ++nt) {
        float v = acc[nt][r] + p1L[nt];
        vals[nt] = v; s += v; q += v * v;
      }
#pragma unroll
      for (int m = 1; m < 16; m <<= 1) { s += __shfl_xor(s, m, 32); q += __shfl_xor(q, m, 32); }
      float mean = s * (1.f / 128.f);
      float var  = q * (1.f / 128.f) - mean * mean;
      float rstd = rsqrtf(fmaxf(var, 0.f) + 1e-5f);
      int row = wave * 32 + mt * 16 + r + rOff;
      _Float16* hr = &sm.AH[row * AH_STRIDE];
#pragma unroll
      for (int nt = 0; nt < 8; ++nt) {
        float v = (vals[nt] - mean) * rstd * g1L[nt] + b1L[nt];
        v = fmaxf(v, 0.f);
        hr[nt * 16 + laneN] = (_Float16)v;
      }
    }
  }
  __syncthreads();

  // ---- GEMM2: h(16x128) @ W2 -> 16x64, LN2, store f32 ----
#pragma unroll
  for (int mt = 0; mt < 2; ++mt) {
    int mrow = wave * 32 + mt * 16 + laneN;
    V16 a[4];
#pragma unroll
    for (int kt = 0; kt < 4; ++kt) {
      const _Float16* base = &sm.AH[mrow * AH_STRIDE + kt * 32 + aKoff];
      a[kt].h[0] = *(const v8h*)base;
      a[kt].h[1] = *(const v8h*)(base + 16);
    }
    v8f acc[4];
#pragma unroll
    for (int nt = 0; nt < 4; ++nt) {
      v8f c = {};
#pragma unroll
      for (int kt = 0; kt < 4; ++kt) {
        int n = nt * 16 + laneN;
        const _Float16* wb = &sm.W2[n * W2_STRIDE + kt * 32 + bKoff];
        V16 bf;
        bf.h[0] = *(const v8h*)wb;
        bf.h[1] = *(const v8h*)(wb + 16);
        c = __builtin_amdgcn_wmma_f32_16x16x32_f16(false, a[kt].v, false, bf.v,
                                                   (short)0, c, false, false);
      }
      acc[nt] = c;
    }
#pragma unroll
    for (int r = 0; r < 8; ++r) {
      float vals[4];
      float s = 0.f, q = 0.f;
#pragma unroll
      for (int nt = 0; nt < 4; ++nt) {
        float v = acc[nt][r] + p2L[nt];
        vals[nt] = v; s += v; q += v * v;
      }
#pragma unroll
      for (int m = 1; m < 16; m <<= 1) { s += __shfl_xor(s, m, 32); q += __shfl_xor(q, m, 32); }
      float mean = s * (1.f / 64.f);
      float var  = q * (1.f / 64.f) - mean * mean;
      float rstd = rsqrtf(fmaxf(var, 0.f) + 1e-5f);
      int row = wave * 32 + mt * 16 + r + rOff;
      int tglob = t0 + row;
      if (tglob < BJ) {
        float* op = out + (size_t)tglob * 64;
#pragma unroll
        for (int nt = 0; nt < 4; ++nt) {
          float v = (vals[nt] - mean) * rstd * g2L[nt] + b2L[nt];
          op[nt * 16 + laneN] = v;
        }
      }
    }
  }
}

extern "C" void kernel_launch(void* const* d_in, const int* in_sizes, int n_in,
                              void* d_out, int out_size, void* d_ws, size_t ws_size,
                              hipStream_t stream) {
  const float* kp    = (const float*)d_in[0];
  const float* pkp   = (const float*)d_in[1];
  const float* je    = (const float*)d_in[2];
  const float* posW  = (const float*)d_in[3];
  const float* posb  = (const float*)d_in[4];
  const float* velW  = (const float*)d_in[5];
  const float* velb  = (const float*)d_in[6];
  // d_in[7] = angvel_W (unused by reference math: angvel_emb is bias-only)
  const float* angb  = (const float*)d_in[8];
  const float* relW  = (const float*)d_in[9];
  const float* relb  = (const float*)d_in[10];
  const float* confW = (const float*)d_in[11];
  const float* confb = (const float*)d_in[12];
  const float* p1W   = (const float*)d_in[13];
  const float* p1b   = (const float*)d_in[14];
  const float* l1g   = (const float*)d_in[15];
  const float* l1b   = (const float*)d_in[16];
  const float* p2W   = (const float*)d_in[17];
  const float* p2b   = (const float*)d_in[18];
  const float* l2g   = (const float*)d_in[19];
  const float* l2b   = (const float*)d_in[20];
  float* out = (float*)d_out;

  int BJ = in_sizes[0] / 3;                 // B * J tokens
  int blocks = (BJ + TPB - 1) / TPB;        // 2176 for B=32768
  jfe_kernel<<<dim3(blocks), dim3(TPB), 0, stream>>>(
      kp, pkp, je, posW, posb, velW, velb, angb, relW, relb, confW, confb,
      p1W, p1b, l1g, l1b, p2W, p2b, l2g, l2b, out, BJ);
}
